// MultiHeadedAttention_10625749090758
// MI455X (gfx1250) — matrix operand order repaired and checked by reading, not compiled, not run on hardware
//
#include <hip/hip_runtime.h>
#include <hip/hip_bf16.h>

// ---------- problem constants ----------
#define Bsz    4
#define Sseq   2048
#define Dmod   1024
#define Hn     16
#define DKv    64            // dk == dv == 64
#define Mrows  (Bsz * Sseq)  // 8192

typedef __attribute__((ext_vector_type(16))) __bf16 v16bf;
typedef __attribute__((ext_vector_type(8)))  float  v8f;

union FragU { v16bf v; uint4 u[2]; };

__device__ __forceinline__ v8f zero_v8f() {
  v8f z;
#pragma unroll
  for (int i = 0; i < 8; ++i) z[i] = 0.0f;
  return z;
}

// A fragment: 16x32 bf16, row-major source with row stride lda (elements).
// ISA layout: lane m=lane%16; VGPR0..3 hold k = kb..kb+7, VGPR4..7 hold k = 16+kb..16+kb+7,
// kb = (lane/16)*8.  Two contiguous 16B chunks per lane.
__device__ __forceinline__ v16bf load_a_frag(const __bf16* A, int lda, int lane) {
  int m  = lane & 15;
  int kb = (lane >> 4) << 3;
  const __bf16* row = A + (size_t)m * lda;
  FragU f;
  f.u[0] = *reinterpret_cast<const uint4*>(row + kb);
  f.u[1] = *reinterpret_cast<const uint4*>(row + 16 + kb);
  return f.v;
}

// B fragment: 32x16 bf16 where element (k,n) = Bt[n*ldb + k] (Bt is N-major, K contiguous).
// ISA layout: lane n=lane%16; elements 0..7 hold k = kb..kb+7, 8..15 hold k = 16+kb..16+kb+7, kb = (lane/16)*8.
__device__ __forceinline__ v16bf load_b_frag(const __bf16* Bt, int ldb, int lane) {
  int n  = lane & 15;
  int kb = (lane >> 4) << 3;
  const __bf16* p = Bt + (size_t)n * ldb;
  FragU f;
  f.u[0] = *reinterpret_cast<const uint4*>(p + kb);
  f.u[1] = *reinterpret_cast<const uint4*>(p + 16 + kb);
  return f.v;
}

__device__ __forceinline__ v8f wmma_bf16(v16bf a, v16bf b, v8f c) {
  // (neg_a, A, neg_b, B, c_mod, C, reuse_a, reuse_b)
  return __builtin_amdgcn_wmma_f32_16x16x32_bf16(false, a, false, b, (short)0, c,
                                                 false, false);
}

// ---------- precision conversion / weight transpose ----------
__global__ void cvt_bf16_kernel(const float* __restrict__ in, __bf16* __restrict__ out, int n) {
  int i = blockIdx.x * blockDim.x + threadIdx.x;
  int stride = gridDim.x * blockDim.x;
  for (; i < n; i += stride) out[i] = (__bf16)in[i];
}

// W: [h][K][N] fp32  ->  Wt: [h][N][K] bf16   (total = h*N*K)
__global__ void transpose_bf16_kernel(const float* __restrict__ W, __bf16* __restrict__ Wt,
                                      int K, int N, int total) {
  int i = blockIdx.x * blockDim.x + threadIdx.x;
  int stride = gridDim.x * blockDim.x;
  int NK = N * K;
  for (; i < total; i += stride) {
    int hh  = i / NK;
    int rem = i - hh * NK;
    int n   = rem / K;
    int k   = rem - n * K;
    Wt[i] = (__bf16)W[(size_t)hh * NK + (size_t)k * N + n];
  }
}

// ---------- QKV projection: per wave a 32x32 tile of (x[8192x1024] * Wt^T) ----------
// grid: x = Mrows/128, y = 2 (n0 = 32*y), z = mat*16 + h.  block = 128 (4 waves).
__global__ void __launch_bounds__(128)
qkv_gemm_kernel(const __bf16* __restrict__ xb,
                const __bf16* __restrict__ wtq,
                const __bf16* __restrict__ wtk,
                const __bf16* __restrict__ wtv,
                __bf16* __restrict__ Qo, __bf16* __restrict__ Ko, __bf16* __restrict__ Vt) {
  int lane = threadIdx.x & 31;
  int wave = threadIdx.x >> 5;
  int m0 = (blockIdx.x * 4 + wave) * 32;
  int n0 = blockIdx.y * 32;
  int mat = blockIdx.z >> 4;  // 0=Q 1=K 2=V
  int h   = blockIdx.z & 15;
  const __bf16* Wt = (mat == 0) ? wtq : (mat == 1) ? wtk : wtv;
  Wt += (size_t)h * (DKv * Dmod);  // [64][1024] bf16, K contiguous

  v8f acc[2][2];
  acc[0][0] = zero_v8f(); acc[0][1] = zero_v8f();
  acc[1][0] = zero_v8f(); acc[1][1] = zero_v8f();

  for (int k = 0; k < Dmod; k += 32) {
    v16bf a0 = load_a_frag(xb + (size_t)m0 * Dmod + k, Dmod, lane);
    v16bf a1 = load_a_frag(xb + (size_t)(m0 + 16) * Dmod + k, Dmod, lane);
    v16bf b0 = load_b_frag(Wt + (size_t)n0 * Dmod + k, Dmod, lane);
    v16bf b1 = load_b_frag(Wt + (size_t)(n0 + 16) * Dmod + k, Dmod, lane);
    acc[0][0] = wmma_bf16(a0, b0, acc[0][0]);
    acc[0][1] = wmma_bf16(a0, b1, acc[0][1]);
    acc[1][0] = wmma_bf16(a1, b0, acc[1][0]);
    acc[1][1] = wmma_bf16(a1, b1, acc[1][1]);
  }

  int nl = lane & 15, half = lane >> 4;
#pragma unroll
  for (int i = 0; i < 2; ++i)
#pragma unroll
    for (int j = 0; j < 2; ++j)
#pragma unroll
      for (int r = 0; r < 8; ++r) {
        int mm = m0 + 16 * i + r + 8 * half;      // global row (b*S + s)
        int n  = n0 + 16 * j + nl;                // 0..63
        int bb = mm >> 11, s = mm & (Sseq - 1);
        __bf16 val = (__bf16)acc[i][j][r];
        if (mat == 0)
          Qo[(((size_t)(bb * Hn + h)) * Sseq + s) * DKv + n] = val;
        else if (mat == 1)
          Ko[(((size_t)(bb * Hn + h)) * Sseq + s) * DKv + n] = val;
        else
          Vt[(((size_t)(bb * Hn + h)) * DKv + n) * Sseq + s] = val;  // V transposed
      }
}

// ---------- flash attention: one wave per 16-query block ----------
// grid = (B*H*S/16)/8 = 1024 blocks, block = 256 (8 waves).
__global__ void __launch_bounds__(256)
flash_attn_kernel(const __bf16* __restrict__ Q, const __bf16* __restrict__ K,
                  const __bf16* __restrict__ Vt, __bf16* __restrict__ AO) {
  __shared__ __align__(16) __bf16 plds[8][16 * 32];  // per-wave 16x32 P tile

  int lane = threadIdx.x & 31;
  int wave = threadIdx.x >> 5;
  int wid  = blockIdx.x * 8 + wave;
  int qb = wid & ((Sseq / 16) - 1);     // query block
  int bh = wid >> 7;                    // b*H + h
  int bb = bh >> 4, h = bh & 15;

  const __bf16* Qp = Q + ((size_t)bh * Sseq + qb * 16) * DKv;
  const __bf16* Kp = K + (size_t)bh * Sseq * DKv;
  const __bf16* Vp = Vt + (size_t)bh * DKv * Sseq;
  __bf16* Op = AO + ((size_t)bb * Sseq + qb * 16) * (Hn * DKv) + h * DKv;

  v16bf qa0 = load_a_frag(Qp, DKv, lane);        // k 0..31
  v16bf qa1 = load_a_frag(Qp + 32, DKv, lane);   // k 32..63

  v8f o0 = zero_v8f(), o1 = zero_v8f(), o2 = zero_v8f(), o3 = zero_v8f();
  float mrow[8], lrow[8];
#pragma unroll
  for (int r = 0; r < 8; ++r) { mrow[r] = -1e30f; lrow[r] = 0.0f; }

  const float scale = 0.125f;  // 1/sqrt(64)
  __bf16* myp = plds[wave];
  int nl = lane & 15, half = lane >> 4;

  for (int t0 = 0; t0 < Sseq; t0 += 32) {
    // scores for 32 keys: two 16x16 tiles, contraction over dk=64 (2 wmma each)
    v8f s0 = zero_v8f(), s1 = zero_v8f();
    s0 = wmma_bf16(qa0, load_b_frag(Kp + (size_t)t0 * DKv, DKv, lane), s0);
    s0 = wmma_bf16(qa1, load_b_frag(Kp + (size_t)t0 * DKv + 32, DKv, lane), s0);
    s1 = wmma_bf16(qa0, load_b_frag(Kp + (size_t)(t0 + 16) * DKv, DKv, lane), s1);
    s1 = wmma_bf16(qa1, load_b_frag(Kp + (size_t)(t0 + 16) * DKv + 32, DKv, lane), s1);

    // online softmax per row (row = r + 8*half; its 16 cols live across one 16-lane half)
#pragma unroll
    for (int r = 0; r < 8; ++r) {
      float x0 = s0[r] * scale, x1 = s1[r] * scale;
      float mx = fmaxf(x0, x1);
#pragma unroll
      for (int off = 1; off < 16; off <<= 1)
        mx = fmaxf(mx, __shfl_xor(mx, off, 32));
      float mnew  = fmaxf(mrow[r], mx);
      float alpha = __expf(mrow[r] - mnew);
      float p0 = __expf(x0 - mnew);
      float p1 = __expf(x1 - mnew);
      float rs = p0 + p1;
#pragma unroll
      for (int off = 1; off < 16; off <<= 1)
        rs += __shfl_xor(rs, off, 32);
      lrow[r] = lrow[r] * alpha + rs;
      mrow[r] = mnew;
      o0[r] *= alpha; o1[r] *= alpha; o2[r] *= alpha; o3[r] *= alpha;
      int row = r + 8 * half;
      myp[row * 32 + nl]      = (__bf16)p0;
      myp[row * 32 + 16 + nl] = (__bf16)p1;
    }
    // same-wave LDS ops are in order; barrier only pins compiler ordering
    __builtin_amdgcn_wave_barrier();
    v16bf pa = load_a_frag(myp, 32, lane);   // P as 16x32 A fragment
    __builtin_amdgcn_wave_barrier();

    // o += P(16x32) * V(32x64): 4 output N-subtiles, V stored transposed [64][S]
    o0 = wmma_bf16(pa, load_b_frag(Vp + (size_t)0  * Sseq + t0, Sseq, lane), o0);
    o1 = wmma_bf16(pa, load_b_frag(Vp + (size_t)16 * Sseq + t0, Sseq, lane), o1);
    o2 = wmma_bf16(pa, load_b_frag(Vp + (size_t)32 * Sseq + t0, Sseq, lane), o2);
    o3 = wmma_bf16(pa, load_b_frag(Vp + (size_t)48 * Sseq + t0, Sseq, lane), o3);
  }

#pragma unroll
  for (int r = 0; r < 8; ++r) {
    float inv = 1.0f / lrow[r];
    int row = r + 8 * half;
    Op[(size_t)row * (Hn * DKv) + nl]      = (__bf16)(o0[r] * inv);
    Op[(size_t)row * (Hn * DKv) + 16 + nl] = (__bf16)(o1[r] * inv);
    Op[(size_t)row * (Hn * DKv) + 32 + nl] = (__bf16)(o2[r] * inv);
    Op[(size_t)row * (Hn * DKv) + 48 + nl] = (__bf16)(o3[r] * inv);
  }
}

// ---------- output projection: [8192x1024]bf16 * WtO -> fp32 out ----------
// grid: x = Mrows/128, y = 32.  block = 128 (4 waves, 32x32 per wave).
__global__ void __launch_bounds__(128)
out_gemm_kernel(const __bf16* __restrict__ A, const __bf16* __restrict__ WtO,
                float* __restrict__ out) {
  int lane = threadIdx.x & 31;
  int wave = threadIdx.x >> 5;
  int m0 = (blockIdx.x * 4 + wave) * 32;
  int n0 = blockIdx.y * 32;

  v8f acc[2][2];
  acc[0][0] = zero_v8f(); acc[0][1] = zero_v8f();
  acc[1][0] = zero_v8f(); acc[1][1] = zero_v8f();

  for (int k = 0; k < Dmod; k += 32) {
    v16bf a0 = load_a_frag(A + (size_t)m0 * Dmod + k, Dmod, lane);
    v16bf a1 = load_a_frag(A + (size_t)(m0 + 16) * Dmod + k, Dmod, lane);
    v16bf b0 = load_b_frag(WtO + (size_t)n0 * Dmod + k, Dmod, lane);
    v16bf b1 = load_b_frag(WtO + (size_t)(n0 + 16) * Dmod + k, Dmod, lane);
    acc[0][0] = wmma_bf16(a0, b0, acc[0][0]);
    acc[0][1] = wmma_bf16(a0, b1, acc[0][1]);
    acc[1][0] = wmma_bf16(a1, b0, acc[1][0]);
    acc[1][1] = wmma_bf16(a1, b1, acc[1][1]);
  }

  int nl = lane & 15, half = lane >> 4;
#pragma unroll
  for (int i = 0; i < 2; ++i)
#pragma unroll
    for (int j = 0; j < 2; ++j)
#pragma unroll
      for (int r = 0; r < 8; ++r) {
        int mm = m0 + 16 * i + r + 8 * half;
        int n  = n0 + 16 * j + nl;
        out[(size_t)mm * Dmod + n] = acc[i][j][r];
      }
}

// ---------- launcher ----------
extern "C" void kernel_launch(void* const* d_in, const int* in_sizes, int n_in,
                              void* d_out, int out_size, void* d_ws, size_t ws_size,
                              hipStream_t stream) {
  const float* x   = (const float*)d_in[0];
  const float* W_Q = (const float*)d_in[1];
  const float* W_K = (const float*)d_in[2];
  const float* W_V = (const float*)d_in[3];
  const float* W_O = (const float*)d_in[4];
  float* out = (float*)d_out;

  char* ws = (char*)d_ws;
  size_t off = 0;
  auto alloc = [&](size_t bytes) -> void* {
    void* p = ws + off;
    off += (bytes + 255) & ~(size_t)255;
    return p;
  };

  const size_t nX  = (size_t)Mrows * Dmod;              // 8,388,608
  const size_t nWh = (size_t)Hn * DKv * Dmod;           // 1,048,576 per proj weight
  const size_t nWO = (size_t)Dmod * Dmod;               // 1,048,576
  const size_t nQ  = (size_t)Bsz * Hn * Sseq * DKv;     // 8,388,608

  __bf16* xb  = (__bf16*)alloc(nX * 2);
  __bf16* wtq = (__bf16*)alloc(nWh * 2);
  __bf16* wtk = (__bf16*)alloc(nWh * 2);
  __bf16* wtv = (__bf16*)alloc(nWh * 2);
  __bf16* wto = (__bf16*)alloc(nWO * 2);
  __bf16* Qb  = (__bf16*)alloc(nQ * 2);
  __bf16* Kb  = (__bf16*)alloc(nQ * 2);
  __bf16* Vtb = (__bf16*)alloc(nQ * 2);
  __bf16* AO  = (__bf16*)alloc(nX * 2);
  (void)ws_size; (void)in_sizes; (void)n_in; (void)out_size;  // ~92.3 MB (88 MiB) used

  cvt_bf16_kernel<<<4096, 256, 0, stream>>>(x, xb, (int)nX);
  transpose_bf16_kernel<<<1024, 256, 0, stream>>>(W_Q, wtq, Dmod, DKv, (int)nWh);
  transpose_bf16_kernel<<<1024, 256, 0, stream>>>(W_K, wtk, Dmod, DKv, (int)nWh);
  transpose_bf16_kernel<<<1024, 256, 0, stream>>>(W_V, wtv, Dmod, DKv, (int)nWh);
  transpose_bf16_kernel<<<1024, 256, 0, stream>>>(W_O, wto, Dmod, Dmod, (int)nWO);

  qkv_gemm_kernel<<<dim3(Mrows / 128, 2, 3 * Hn), 128, 0, stream>>>(
      xb, wtq, wtk, wtv, Qb, Kb, Vtb);

  flash_attn_kernel<<<(Bsz * Hn * (Sseq / 16)) / 8, 256, 0, stream>>>(Qb, Kb, Vtb, AO);

  out_gemm_kernel<<<dim3(Mrows / 128, Dmod / 32), 128, 0, stream>>>(AO, wto, out);
}
